// CrossAttention_28767690949003
// MI455X (gfx1250) — compile-verified, matrix operand order checked by reading, not run on hardware
//
#include <hip/hip_runtime.h>
#include <hip/hip_bf16.h>

typedef __attribute__((ext_vector_type(16))) _Float16 v16h;
typedef __attribute__((ext_vector_type(8)))  _Float16 v8h;
typedef __attribute__((ext_vector_type(8)))  float    v8f;
typedef __attribute__((ext_vector_type(4)))  float    v4f;
typedef __attribute__((ext_vector_type(4)))  unsigned u32x4;
typedef __attribute__((ext_vector_type(8)))  int      i32x8;
typedef __attribute__((ext_vector_type(4)))  int      i32x4;

#define B_SZ  2
#define QL    4096
#define KL    4096
#define QD    512
#define CD    768
#define NH    8
#define DHD   64
#define INNER 512
#define SCALE 0.125f /* 64^-0.5 */

// ---------------------------------------------------------------------------
// CDNA5 16-bit WMMA fragment layout (CDNA5 ISA §7.12.2):
// A (16x32): lane holds row M=lane%16. Element j=0..7 -> K = 8*hi + j,
// j=8..15 -> K = 16 + 8*hi + (j-8), hi = lane/16.  => TWO CONTIGUOUS 16B
// chunks per lane.  B fragments from a row-major N x K (transposed) source
// follow the identical pattern with the row selected by N=lane%16.
// ---------------------------------------------------------------------------
__device__ __forceinline__ v16h frag_cat(v8h lo, v8h hh) {
  return __builtin_shufflevector(lo, hh, 0, 1, 2, 3, 4, 5, 6, 7,
                                 8, 9, 10, 11, 12, 13, 14, 15);
}

__device__ __forceinline__ v8f wmma16(v16h a, v16h b, v8f c) {
  // (neg_a, A, neg_b, B, c_mod, C, reuse_a, reuse_b)
  return __builtin_amdgcn_wmma_f32_16x16x32_f16(false, a, false, b, (short)0, c,
                                                false, false);
}

// LDS A fragment: rows at base + (lane%16)*ld, cols k0..k0+31 -> 2x ds b128
__device__ __forceinline__ v16h ldsA(const _Float16* base, int ld, int k0, int lane) {
  int ln = lane & 15, hi = lane >> 4;
  const _Float16* p = base + ln * ld + k0 + 8 * hi;
  return frag_cat(*(const v8h*)p, *(const v8h*)(p + 16));
}
// LDS B fragment from transposed (row-major N x K) source -> 2x ds b128
__device__ __forceinline__ v16h ldsBT(const _Float16* base, int ld, int k0, int n0, int lane) {
  int ln = lane & 15, hi = lane >> 4;
  const _Float16* p = base + (n0 + ln) * ld + k0 + 8 * hi;
  return frag_cat(*(const v8h*)p, *(const v8h*)(p + 16));
}
// Global A fragment from f32 source -> 4x global b128 + packed cvt
__device__ __forceinline__ v16h gA_f32(const float* X, int ld, int m0, int k0, int lane) {
  int ln = lane & 15, hi = lane >> 4;
  const float* p = X + (size_t)(m0 + ln) * ld + k0 + 8 * hi;
  v4f c0 = *(const v4f*)p, c1 = *(const v4f*)(p + 4);
  v4f c2 = *(const v4f*)(p + 16), c3 = *(const v4f*)(p + 20);
  v16h a;
#pragma unroll
  for (int j = 0; j < 4; ++j) {
    a[j]      = (_Float16)c0[j];
    a[j + 4]  = (_Float16)c1[j];
    a[j + 8]  = (_Float16)c2[j];
    a[j + 12] = (_Float16)c3[j];
  }
  return a;
}
// Global A fragment from f16 source -> 2x global b128
__device__ __forceinline__ v16h gA_f16(const _Float16* X, int ld, int m0, int k0, int lane) {
  int ln = lane & 15, hi = lane >> 4;
  const _Float16* p = X + (size_t)(m0 + ln) * ld + k0 + 8 * hi;
  return frag_cat(*(const v8h*)p, *(const v8h*)(p + 16));
}
// Global B fragment from pre-transposed f16 weights Wt[n][k] -> 2x global b128
__device__ __forceinline__ v16h gBT_f16(const _Float16* Wt, int ld, int k0, int n0, int lane) {
  int ln = lane & 15, hi = lane >> 4;
  const _Float16* p = Wt + (size_t)(n0 + ln) * ld + k0 + 8 * hi;
  return frag_cat(*(const v8h*)p, *(const v8h*)(p + 16));
}

// ---------------------------------------------------------------------------
// One-shot weight transpose+convert: Wt(f16)[N][Kd] = W(f32)[Kd][N]^T
// ---------------------------------------------------------------------------
__global__ void w_transpose(const float* __restrict__ W, _Float16* __restrict__ Wt,
                            int Kd, int N) {
  int idx = blockIdx.x * 256 + threadIdx.x;
  int k = idx / N, n = idx % N;  // coalesced read along n
  Wt[(size_t)n * Kd + k] = (_Float16)W[idx];
}

// ---------------------------------------------------------------------------
// Projection GEMM: Y(f16)[M x N] = X(f32)[M x Kd] * Wt(f16)[N x Kd]^T
// block = 128 (4 waves), wave owns one 16x16 tile; grid = (M/16, N/64)
// ---------------------------------------------------------------------------
__global__ void proj_f32_to_f16(const float* __restrict__ X, const _Float16* __restrict__ Wt,
                                _Float16* __restrict__ Y, int Kd, int N) {
  int lane = threadIdx.x & 31, wv = threadIdx.x >> 5;
  int m0 = blockIdx.x * 16;
  int n0 = (blockIdx.y * 4 + wv) * 16;
  v8f acc = {};
  for (int k0 = 0; k0 < Kd; k0 += 32) {
    v16h a = gA_f32(X, Kd, m0, k0, lane);
    v16h b = gBT_f16(Wt, Kd, k0, n0, lane);
    acc = wmma16(a, b, acc);
  }
  int ln = lane & 15, hi = lane >> 4;
#pragma unroll
  for (int r = 0; r < 8; ++r)
    Y[(size_t)(m0 + r + 8 * hi) * N + n0 + ln] = (_Float16)acc[r];
}

// ---------------------------------------------------------------------------
// Flash attention: grid = (B*H, QL/128), block = 256 (8 waves).
// 128 Q rows x DH=64 per block; each wave owns 16 Q rows. K tile staged by
// the Tensor Data Mover (TDM) when available; V staged TRANSPOSED in LDS so
// both GEMMs use contiguous b128 fragment loads.
// ---------------------------------------------------------------------------
__global__ void flash_attn(const _Float16* __restrict__ Qh, const _Float16* __restrict__ Kh,
                           const _Float16* __restrict__ Vh, _Float16* __restrict__ Oh) {
  __shared__ __align__(16) _Float16 sQ[128 * DHD];   // 16 KB
  __shared__ __align__(16) _Float16 sK[64 * DHD];    // 8 KB   [key][d]
  __shared__ __align__(16) _Float16 sVt[DHD * 64];   // 8 KB   [d][key]
  __shared__ __align__(16) _Float16 sP[8][16 * 64];  // 16 KB  wave-private P

  int tid = threadIdx.x, lane = tid & 31, wv = tid >> 5;
  int b = blockIdx.x >> 3, h = blockIdx.x & 7;
  int q0 = blockIdx.y * 128;
  int ln = lane & 15, hi = lane >> 4;

  // stage Q tile (128 x 64 f16; rows are 128B-contiguous in Qh)
  for (int i = tid; i < 128 * 8; i += 256) {
    int row = i >> 3, c = i & 7;
    ((uint4*)sQ)[i] =
        ((const uint4*)(Qh + (size_t)(b * QL + q0 + row) * INNER + h * DHD))[c];
  }
  __syncthreads();
  // Q A-fragments are kt-invariant: hoist into registers
  const _Float16* qbase = sQ + wv * 16 * DHD;
  v16h qfrag0 = ldsA(qbase, DHD, 0, lane);
  v16h qfrag1 = ldsA(qbase, DHD, 32, lane);

  v8f o[4] = {};
  float m_run[8], l_run[8];
#pragma unroll
  for (int r = 0; r < 8; ++r) { m_run[r] = -3.0e38f; l_run[r] = 0.f; }

  _Float16* pbuf = sP[wv];

  for (int kt = 0; kt < KL / 64; ++kt) {
    __syncthreads();  // previous iteration's compute done before restaging

#if __has_builtin(__builtin_amdgcn_tensor_load_to_lds) && \
    __has_builtin(__builtin_amdgcn_s_wait_tensorcnt)
    // ---- K tile via TDM: 2D tile 64(d) x 64(keys), f16, row stride 512 ----
    // D# built per CDNA5 ISA.
    if (wv == 0) {
      unsigned lds_off = (unsigned)(size_t)(void*)sK;  // addr[31:0] = LDS offset
      unsigned long long ga =
          (unsigned long long)(const void*)(Kh + (size_t)(b * KL + kt * 64) * INNER + h * DHD);
      u32x4 g0;
      g0[0] = 1u;                                   // count=1 (valid user D#)
      g0[1] = lds_off;                              // lds_addr (bytes)
      g0[2] = (unsigned)(ga & 0xFFFFFFFFull);       // global_addr[31:0]
      g0[3] = (unsigned)((ga >> 32) & 0x01FFFFFFull) | (2u << 30);  // [56:32]|type=2
      i32x8 g1;
      g1[0] = 1 << 16;       // wg_mask=0 (not in cluster), data_size=1 (2B)
      g1[1] = 64 << 16;      // tensor_dim0[15:0]=64 in bits[31:16]
      g1[2] = 64 << 16;      // tensor_dim0 hi=0 | tensor_dim1[15:0]=64
      g1[3] = 64 << 16;      // tensor_dim1 hi=0 | tile_dim0=64
      g1[4] = 64;            // tile_dim1=64, tile_dim2=0
      g1[5] = INNER;         // tensor_dim0_stride[31:0]=512 elements
      g1[6] = 0;             // stride0 hi | stride1 lo
      g1[7] = 0;             // stride1 hi
      i32x4 z4 = {0, 0, 0, 0};                       // 2D tensor: groups 2/3 unused
      i32x8 z8 = {0, 0, 0, 0, 0, 0, 0, 0};           // 6-arg variant: extra group
      __builtin_amdgcn_tensor_load_to_lds(g0, g1, z4, z4, z8, 0);
    }
#else
    // ---- fallback: manual K tile staging (64 x 8 uint4) ----
    for (int i = tid; i < 64 * 8; i += 256) {
      int row = i >> 3, c = i & 7;
      ((uint4*)sK)[i] =
          ((const uint4*)(Kh + (size_t)(b * KL + kt * 64 + row) * INNER + h * DHD))[c];
    }
#endif

    // ---- V tile transposed (overlaps with TDM): key-pair x 8 d's, b32 packs
    {
      int p = tid & 31, dc = tid >> 5;
      size_t g0v = (size_t)(b * KL + kt * 64 + 2 * p) * INNER + h * DHD + dc * 8;
      uint4 r0 = *(const uint4*)(Vh + g0v);
      uint4 r1 = *(const uint4*)(Vh + g0v + INNER);
      const _Float16* x0 = (const _Float16*)&r0;
      const _Float16* x1 = (const _Float16*)&r1;
#pragma unroll
      for (int dd = 0; dd < 8; ++dd) {
        int d = dc * 8 + dd;
        union { _Float16 h2[2]; unsigned u; } pk;
        pk.h2[0] = x0[dd];
        pk.h2[1] = x1[dd];
        *(unsigned*)(sVt + d * 64 + 2 * p) = pk.u;
      }
    }

#if __has_builtin(__builtin_amdgcn_tensor_load_to_lds) && \
    __has_builtin(__builtin_amdgcn_s_wait_tensorcnt)
    if (wv == 0) __builtin_amdgcn_s_wait_tensorcnt((short)0);  // TDM done
#endif
    __syncthreads();

    if (kt + 1 < KL / 64) {  // gfx1250 global_prefetch_b8 for next V tile
      size_t g = (size_t)(b * KL + (kt + 1) * 64 + (tid >> 2)) * INNER + h * DHD;
      __builtin_prefetch(Vh + g, 0, 0);
    }

    // ---- S = Q * K^T  (16 rows x 64 keys per wave) ----
    v8f s[4] = {};
#pragma unroll
    for (int t = 0; t < 4; ++t) {
      s[t] = wmma16(qfrag0, ldsBT(sK, DHD, 0, t * 16, lane), s[t]);
      s[t] = wmma16(qfrag1, ldsBT(sK, DHD, 32, t * 16, lane), s[t]);
    }

    // ---- scaled online softmax (per-row stats in 16-lane shfl groups) ----
    float mnew[8], lsum[8];
#pragma unroll
    for (int r = 0; r < 8; ++r) {
      float v = fmaxf(fmaxf(s[0][r], s[1][r]), fmaxf(s[2][r], s[3][r])) * SCALE;
#pragma unroll
      for (int msk = 1; msk <= 8; msk <<= 1) v = fmaxf(v, __shfl_xor(v, msk, 32));
      mnew[r] = fmaxf(m_run[r], v);
      lsum[r] = 0.f;
    }
#pragma unroll
    for (int t = 0; t < 4; ++t)
#pragma unroll
      for (int r = 0; r < 8; ++r) {
        float p = __expf(s[t][r] * SCALE - mnew[r]);
        s[t][r] = p;
        lsum[r] += p;
      }
#pragma unroll
    for (int r = 0; r < 8; ++r) {
#pragma unroll
      for (int msk = 1; msk <= 8; msk <<= 1) lsum[r] += __shfl_xor(lsum[r], msk, 32);
      float corr = __expf(m_run[r] - mnew[r]);
      l_run[r] = l_run[r] * corr + lsum[r];
      m_run[r] = mnew[r];
#pragma unroll
      for (int t = 0; t < 4; ++t) o[t][r] *= corr;
    }

    // ---- stage P row-major in wave-private LDS (C->A layout round trip;
    //      LDS ops are in-order within a wave, no block barrier needed) ----
#pragma unroll
    for (int t = 0; t < 4; ++t)
#pragma unroll
      for (int r = 0; r < 8; ++r)
        pbuf[(r + 8 * hi) * 64 + t * 16 + ln] = (_Float16)s[t][r];

    // ---- O += P * V  (B fragments from transposed sVt: contiguous b128) ----
#pragma unroll
    for (int ks = 0; ks < 2; ++ks) {
      v16h a = ldsA(pbuf, 64, ks * 32, lane);
#pragma unroll
      for (int t = 0; t < 4; ++t)
        o[t] = wmma16(a, ldsBT(sVt, 64, ks * 32, t * 16, lane), o[t]);
    }
  }

  // ---- normalize and write attention output (f16, [B*QL, INNER]) ----
#pragma unroll
  for (int t = 0; t < 4; ++t)
#pragma unroll
    for (int r = 0; r < 8; ++r) {
      int row = q0 + wv * 16 + r + 8 * hi;
      Oh[(size_t)(b * QL + row) * INNER + h * DHD + t * 16 + ln] =
          (_Float16)(o[t][r] / l_run[r]);
    }
}

// ---------------------------------------------------------------------------
// Output projection: out(f32)[M x QD] = Oh(f16)[M x INNER] * WoT(f16)^T + bo
// ---------------------------------------------------------------------------
__global__ void out_proj(const _Float16* __restrict__ Oh, const _Float16* __restrict__ WoT,
                         const float* __restrict__ bo, float* __restrict__ out) {
  int lane = threadIdx.x & 31, wv = threadIdx.x >> 5;
  int m0 = blockIdx.x * 16;
  int n0 = (blockIdx.y * 4 + wv) * 16;
  v8f acc = {};
  for (int k0 = 0; k0 < INNER; k0 += 32) {
    v16h a = gA_f16(Oh, INNER, m0, k0, lane);
    v16h b = gBT_f16(WoT, INNER, k0, n0, lane);
    acc = wmma16(a, b, acc);
  }
  int ln = lane & 15, hi = lane >> 4;
  float bias = bo[n0 + ln];
#pragma unroll
  for (int r = 0; r < 8; ++r)
    out[(size_t)(m0 + r + 8 * hi) * QD + n0 + ln] = acc[r] + bias;
}

// ---------------------------------------------------------------------------
extern "C" void kernel_launch(void* const* d_in, const int* in_sizes, int n_in,
                              void* d_out, int out_size, void* d_ws, size_t ws_size,
                              hipStream_t stream) {
  const float* x   = (const float*)d_in[0];
  const float* ctx = (const float*)d_in[1];
  const float* Wq  = (const float*)d_in[2];
  const float* Wk  = (const float*)d_in[3];
  const float* Wv  = (const float*)d_in[4];
  const float* Wo  = (const float*)d_in[5];
  const float* bo  = (const float*)d_in[6];
  float* out = (float*)d_out;

  const size_t M = (size_t)B_SZ * QL;  // 8192 (== B*KL)
  _Float16* Qh  = (_Float16*)d_ws;     // 8 MB each
  _Float16* Kh  = Qh + M * INNER;
  _Float16* Vh  = Kh + M * INNER;
  _Float16* Oh  = Vh + M * INNER;
  _Float16* WqT = Oh + M * INNER;            // [INNER][QD]
  _Float16* WkT = WqT + (size_t)INNER * QD;  // [INNER][CD]
  _Float16* WvT = WkT + (size_t)INNER * CD;  // [INNER][CD]
  _Float16* WoT = WvT + (size_t)INNER * CD;  // [QD][INNER]

  // weight transpose+convert (tiny; L2-resident afterwards)
  w_transpose<<<(QD * INNER) / 256, 256, 0, stream>>>(Wq, WqT, QD, INNER);
  w_transpose<<<(CD * INNER) / 256, 256, 0, stream>>>(Wk, WkT, CD, INNER);
  w_transpose<<<(CD * INNER) / 256, 256, 0, stream>>>(Wv, WvT, CD, INNER);
  w_transpose<<<(INNER * QD) / 256, 256, 0, stream>>>(Wo, WoT, INNER, QD);

  dim3 gproj((unsigned)(M / 16), INNER / 64);  // (512, 8)
  proj_f32_to_f16<<<gproj, 128, 0, stream>>>(x,   WqT, Qh, QD, INNER);
  proj_f32_to_f16<<<gproj, 128, 0, stream>>>(ctx, WkT, Kh, CD, INNER);
  proj_f32_to_f16<<<gproj, 128, 0, stream>>>(ctx, WvT, Vh, CD, INNER);

  flash_attn<<<dim3(B_SZ * NH, QL / 128), 256, 0, stream>>>(Qh, Kh, Vh, Oh);

  dim3 gout((unsigned)(M / 16), QD / 64);
  out_proj<<<gout, 128, 0, stream>>>(Oh, WoT, bo, out);
}
